// lightST_43198781063352
// MI455X (gfx1250) — matrix operand order repaired and checked by reading, not compiled, not run on hardware
//
#include <hip/hip_runtime.h>
#include <math.h>

#define N_CELLS 20000
#define N_GENES 2000
#define DIM     64
#define E_EXP   2000000
#define E_ADJ   120000
#define E_POS   500000
#define E_NEG   500000
#define NCLUST  7

typedef __attribute__((ext_vector_type(16))) _Float16 v16h;
typedef __attribute__((ext_vector_type(8)))  float    v8f;

// ---------------------------------------------------------------- zero scratch
__global__ void k_zero(float* __restrict__ p, int n) {
    int i = blockIdx.x * blockDim.x + threadIdx.x;
    int stride = gridDim.x * blockDim.x;
    for (; i < n; i += stride) p[i] = 0.0f;
}

// ------------------------------------------- message passing: gather + scatter
// acc[to[e]] += feat[from[e]] * cj[from[e]]   (16 lanes x float4 per edge)
__global__ void k_mp(const float* __restrict__ feat,
                     const int*   __restrict__ from,
                     const int*   __restrict__ to,
                     const float* __restrict__ cj,
                     float*       __restrict__ acc,
                     int n_edges) {
    int id = blockIdx.x * blockDim.x + threadIdx.x;
    if (id >= n_edges * 16) return;
    int e  = id >> 4;
    int ch = (id & 15) << 2;
    int s  = from[e];
    int d  = to[e];
    float c = cj[s];
    const float4 v = *(const float4*)(feat + (size_t)s * DIM + ch);
    float* o = acc + (size_t)d * DIM + ch;
    atomicAdd(o + 0, v.x * c);
    atomicAdd(o + 1, v.y * c);
    atomicAdd(o + 2, v.z * c);
    atomicAdd(o + 3, v.w * c);
}

// row-wise scale by ci
__global__ void k_scale(float* __restrict__ buf, const float* __restrict__ ci, int rows) {
    int i = blockIdx.x * blockDim.x + threadIdx.x;
    if (i >= rows * DIM) return;
    buf[i] *= ci[i >> 6];
}

// out = w[0]*f0 + w[1]*f1 + w[2]*f2
__global__ void k_combine(float* __restrict__ out,
                          const float* __restrict__ f0,
                          const float* __restrict__ f1,
                          const float* __restrict__ f2,
                          const float* __restrict__ w, int n) {
    int i = blockIdx.x * blockDim.x + threadIdx.x;
    if (i >= n) return;
    out[i] = w[0] * f0[i] + w[1] * f1[i] + w[2] * f2[i];
}

// acc[row[e]] += val[e] * x[col[e]]
__global__ void k_spmm(const float* __restrict__ x,
                       const int*   __restrict__ row,
                       const int*   __restrict__ col,
                       const float* __restrict__ val,
                       float*       __restrict__ acc,
                       int n_edges) {
    int id = blockIdx.x * blockDim.x + threadIdx.x;
    if (id >= n_edges * 16) return;
    int e  = id >> 4;
    int ch = (id & 15) << 2;
    int r  = row[e];
    int c  = col[e];
    float v = val[e];
    const float4 xv = *(const float4*)(x + (size_t)c * DIM + ch);
    float* o = acc + (size_t)r * DIM + ch;
    atomicAdd(o + 0, v * xv.x);
    atomicAdd(o + 1, v * xv.y);
    atomicAdd(o + 2, v * xv.z);
    atomicAdd(o + 3, v * xv.w);
}

// --------------------------------------------------- q via WMMA cross-term
// One wave32 per 16-row tile of c_z. dot = c_z(16x64) * cluster^T(64x16, 7 real
// cols zero-padded), computed as two v_wmma_f32_16x16x32_f16 (K = 0..31, 32..63).
// d2 = |c|^2 - 2*dot + |mu|^2 ; q = (1/(1+d2)) normalized (ALPHA=1 -> exponent 1).
__global__ void __launch_bounds__(32) k_q(const float* __restrict__ cz,
                                          const float* __restrict__ mu,
                                          float* __restrict__ outq) {
    __shared__ float dots[16][16];
    const int lane = threadIdx.x;          // 0..31, wave32, EXEC all ones
    const int tile = blockIdx.x;           // 20000/16 = 1250 tiles exactly
    const int rrow = tile * 16 + (lane & 15);
    const int kbase = (lane < 16) ? 0 : 16; // A layout: lanes 0-15 K 0..15
    const float* crow = cz + (size_t)rrow * DIM;

    v16h a0, a1;
    #pragma unroll
    for (int j = 0; j < 8; ++j) {
        a0[j]     = (_Float16)crow[kbase + j];
        a0[8 + j] = (_Float16)crow[kbase + 8 + j];
        a1[j]     = (_Float16)crow[kbase + 32 + j];
        a1[8 + j] = (_Float16)crow[kbase + 40 + j];
    }

    const int n    = lane & 15;            // B/C column
    const int koff = (lane < 16) ? 0 : 16; // B layout: lanes 0-15 K 0..15
    v16h b0, b1;
    #pragma unroll
    for (int j = 0; j < 16; ++j) {
        float bv0 = (n < NCLUST) ? mu[n * DIM + koff + j]      : 0.0f;
        float bv1 = (n < NCLUST) ? mu[n * DIM + 32 + koff + j] : 0.0f;
        b0[j] = (_Float16)bv0;
        b1[j] = (_Float16)bv1;
    }

    v8f acc = {};
    acc = __builtin_amdgcn_wmma_f32_16x16x32_f16(false, a0, false, b0,
                                                 (short)0, acc, false, false);
    acc = __builtin_amdgcn_wmma_f32_16x16x32_f16(false, a1, false, b1,
                                                 (short)0, acc, false, false);

    // C/D layout: VGPR r -> M = r + (lane<16 ? 0 : 8), N = lane&15
    const int rbase = (lane < 16) ? 0 : 8;
    #pragma unroll
    for (int r = 0; r < 8; ++r) dots[rbase + r][n] = acc[r];
    __syncthreads();

    if (lane < 16) {
        const int grow = tile * 16 + lane;
        const float* cr = cz + (size_t)grow * DIM;
        float cn = 0.0f;
        for (int d = 0; d < DIM; ++d) cn += cr[d] * cr[d];
        float q[NCLUST];
        float qs = 0.0f;
        #pragma unroll
        for (int k = 0; k < NCLUST; ++k) {
            float mn = 0.0f;
            for (int d = 0; d < DIM; ++d) { float m = mu[k * DIM + d]; mn += m * m; }
            float d2 = cn - 2.0f * dots[lane][k] + mn;
            float qq = 1.0f / (1.0f + d2);   // ALPHA = 1
            q[k] = qq;
            qs += qq;
        }
        float inv = 1.0f / qs;
        #pragma unroll
        for (int k = 0; k < NCLUST; ++k)
            outq[(size_t)grow * NCLUST + k] = q[k] * inv;
    }
}

// ------------------------------------------- edge dot + sigmoid (16 lanes/edge)
__global__ void k_edge(const float* __restrict__ cz,
                       const float* __restrict__ gz,
                       const int*   __restrict__ u,
                       const int*   __restrict__ v,
                       float*       __restrict__ out, int n) {
    int id = blockIdx.x * blockDim.x + threadIdx.x;
    int e  = id >> 4;
    if (e >= n) return;
    int l  = id & 15;
    int uu = u[e];
    int vv = v[e];
    const float4 a = *(const float4*)(cz + (size_t)uu * DIM + l * 4);
    const float4 b = *(const float4*)(gz + (size_t)vv * DIM + l * 4);
    float p = a.x * b.x + a.y * b.y + a.z * b.z + a.w * b.w;
    p += __shfl_xor(p, 8, 16);
    p += __shfl_xor(p, 4, 16);
    p += __shfl_xor(p, 2, 16);
    p += __shfl_xor(p, 1, 16);
    if (l == 0) out[e] = 1.0f / (1.0f + __expf(-p));
}

// ---------------------------------------------------------------------- driver
extern "C" void kernel_launch(void* const* d_in, const int* in_sizes, int n_in,
                              void* d_out, int out_size, void* d_ws, size_t ws_size,
                              hipStream_t stream) {
    (void)in_sizes; (void)n_in; (void)out_size; (void)ws_size;

    const float* cell    = (const float*)d_in[0];
    const float* gene    = (const float*)d_in[1];
    const float* clus    = (const float*)d_in[2];
    const float* wts     = (const float*)d_in[3];
    const float* cj_cell = (const float*)d_in[4];
    const float* ci_cell = (const float*)d_in[5];
    const float* cj_gene = (const float*)d_in[6];
    const float* ci_gene = (const float*)d_in[7];
    const float* adj_val = (const float*)d_in[8];
    const int*   exp_src = (const int*)d_in[9];
    const int*   exp_dst = (const int*)d_in[10];
    const int*   adj_row = (const int*)d_in[11];
    const int*   adj_col = (const int*)d_in[12];
    const int*   pos_u   = (const int*)d_in[13];
    const int*   pos_v   = (const int*)d_in[14];
    const int*   neg_u   = (const int*)d_in[15];
    const int*   neg_v   = (const int*)d_in[16];
    float* out = (float*)d_out;

    // workspace layout (floats); accumulated buffers first so one zero pass covers them
    float* ws  = (float*)d_ws;
    float* c1  = ws;
    float* c2  = c1  + N_CELLS * DIM;
    float* ch2 = c2  + N_CELLS * DIM;
    float* cz  = ch2 + N_CELLS * DIM;
    float* g1  = cz  + N_CELLS * DIM;
    float* g2  = g1  + N_GENES * DIM;
    float* ch1 = g2  + N_GENES * DIM;   // written directly, no zero needed
    float* gz  = ch1 + N_CELLS * DIM;   // written directly, no zero needed

    const int B = 256;
    const int zeroN = 4 * N_CELLS * DIM + 2 * N_GENES * DIM;
    k_zero<<<(zeroN + B - 1) / B, B, 0, stream>>>(ws, zeroN);

    const int mpW = E_EXP * 16;
    const int mpG = (mpW + B - 1) / B;

    // layer 1
    k_mp<<<mpG, B, 0, stream>>>(cell, exp_src, exp_dst, cj_cell, g1, E_EXP);
    k_mp<<<mpG, B, 0, stream>>>(gene, exp_dst, exp_src, cj_gene, c1, E_EXP);
    k_scale<<<(N_GENES * DIM + B - 1) / B, B, 0, stream>>>(g1, ci_gene, N_GENES);
    k_scale<<<(N_CELLS * DIM + B - 1) / B, B, 0, stream>>>(c1, ci_cell, N_CELLS);

    // layer 2
    k_mp<<<mpG, B, 0, stream>>>(c1, exp_src, exp_dst, cj_cell, g2, E_EXP);
    k_mp<<<mpG, B, 0, stream>>>(g1, exp_dst, exp_src, cj_gene, c2, E_EXP);
    k_scale<<<(N_GENES * DIM + B - 1) / B, B, 0, stream>>>(g2, ci_gene, N_GENES);
    k_scale<<<(N_CELLS * DIM + B - 1) / B, B, 0, stream>>>(c2, ci_cell, N_CELLS);

    // weighted combines
    k_combine<<<(N_CELLS * DIM + B - 1) / B, B, 0, stream>>>(ch1, cell, c1, c2, wts, N_CELLS * DIM);
    k_combine<<<(N_GENES * DIM + B - 1) / B, B, 0, stream>>>(gz, gene, g1, g2, wts, N_GENES * DIM);

    // two SpMM passes
    const int spW = E_ADJ * 16;
    const int spG = (spW + B - 1) / B;
    k_spmm<<<spG, B, 0, stream>>>(ch1, adj_row, adj_col, adj_val, ch2, E_ADJ);
    k_spmm<<<spG, B, 0, stream>>>(ch2, adj_row, adj_col, adj_val, cz, E_ADJ);

    // soft assignment q (WMMA), one wave per 16-row tile
    k_q<<<N_CELLS / 16, 32, 0, stream>>>(cz, clus, out + E_POS + E_NEG);

    // pos / neg edge predictions
    k_edge<<<(E_POS * 16 + B - 1) / B, B, 0, stream>>>(cz, gz, pos_u, pos_v, out, E_POS);
    k_edge<<<(E_NEG * 16 + B - 1) / B, B, 0, stream>>>(cz, gz, neg_u, neg_v, out + E_POS, E_NEG);
}
